// Absolute_attention_89902255440854
// MI455X (gfx1250) — compile-verified, matrix operand order checked by reading, not run on hardware
//
#include <hip/hip_runtime.h>
#include <hip/hip_bf16.h>

// ---------------------------------------------------------------------------
// MI455X (gfx1250) fused "absolute attention":
//   K1: per (64-token, head) tile -> WMMA bf16 GEMM for [Wq|Wk|Wv] slices,
//       softmax/gate epilogue, writes g (bf16) to workspace.
//   K2: out = g @ Wo^T + bo, WMMA bf16 GEMM; G tile staged by the Tensor
//       Data Mover (TENSOR_LOAD_TO_LDS) when the toolchain exposes it.
// ---------------------------------------------------------------------------

typedef __attribute__((ext_vector_type(16))) __bf16        v16bf;
typedef __attribute__((ext_vector_type(8)))  float         v8f;
typedef __attribute__((ext_vector_type(4)))  unsigned int  v4u;
typedef __attribute__((ext_vector_type(2)))  unsigned int  v2u;
typedef __attribute__((ext_vector_type(8)))  int           v8i;
typedef __attribute__((ext_vector_type(4)))  int           v4i;

union FragU { v16bf v; v4u q[2]; };

__device__ __forceinline__ unsigned short f32_bf16(float f) {
  unsigned int u = __builtin_bit_cast(unsigned int, f);
  u += 0x7FFFu + ((u >> 16) & 1u);           // round-to-nearest-even
  return (unsigned short)(u >> 16);
}

// Convert float4 -> 4 packed bf16 and store as one 8-byte LDS write.
__device__ __forceinline__ void store_bf16x4(unsigned short* d, float4 v) {
  v2u u;
  u.x = (unsigned int)f32_bf16(v.x) | ((unsigned int)f32_bf16(v.y) << 16);
  u.y = (unsigned int)f32_bf16(v.z) | ((unsigned int)f32_bf16(v.w) << 16);
  *(v2u*)d = u;
}

// Load one 16x32 bf16 fragment from LDS (row stride 72 bf16 = 144 B).
// ISA 16-bit A/B layout: lane(0..15)=row, hi=lane>>4 selects K-half; per lane
// the 16 elements are two contiguous 16B chunks -> two ds_load_b128.
__device__ __forceinline__ v16bf load_frag(const unsigned short* base, int lane) {
  const unsigned short* row = base + (lane & 15) * 72 + (lane >> 4) * 8;
  FragU f;
  f.q[0] = *(const v4u*)(row);
  f.q[1] = *(const v4u*)(row + 16);
  return f.v;
}

// ---------------------------------------------------------------------------
// Kernel 1: projections + softmax gate.  grid = (tokens/64)*heads = 4096 WGs.
// N layout in Ws/Zs: [0,65) q-logits (pad to 80), [80,144) k-logits,
// [144,208) v, [208,224) zero pad.  14 N-tiles of 16; 8 waves split as
// (M-tile 0..3) x (N-half 0/1), exactly 7 unconditional WMMA tiles per wave.
// ---------------------------------------------------------------------------
__global__ __launch_bounds__(256) void proj_gate_kernel(
    const float* __restrict__ x, const float* __restrict__ amask,
    const float* __restrict__ Wq, const float* __restrict__ Wk,
    const float* __restrict__ Wv, const float* __restrict__ ta,
    const float* __restrict__ hdelt, unsigned short* __restrict__ G) {
  union Smem {
    struct { unsigned short Xs[64][72]; unsigned short Ws[224][72]; } l;  // 41.5 KB
    float Zs[64][228];                                                    // 58.4 KB
  };
  __shared__ Smem sm;

  const int tid  = threadIdx.x;
  const int lane = tid & 31;
  const int wv   = tid >> 5;
  const int head      = blockIdx.x & 15;
  const int tokenBase = (int)(blockIdx.x >> 4) * 64;

  const int mtile = wv >> 1;            // 0..3  (16 tokens each)
  const int tBeg  = (wv & 1) * 7;       // N-tiles [0,7) or [7,14)

  v8f acc[7] = {};

  const int cg = tid & 15;              // float4 group within a 64-wide row
  const int rr = tid >> 4;              // 0..15

  for (int k0 = 0; k0 < 1024; k0 += 64) {
    // ---- stage X tile (64x64), f32 -> bf16
#pragma unroll
    for (int p = 0; p < 4; ++p) {
      const int row = rr + p * 16;
      const float4 v = *(const float4*)(x + (size_t)(tokenBase + row) * 1024 + k0 + cg * 4);
      store_bf16x4(&sm.l.Xs[row][cg * 4], v);
    }
    // ---- stage W tile (224x64): [Wq 65 | pad | Wk 64 | Wv 64 | pad]
#pragma unroll
    for (int p = 0; p < 14; ++p) {
      const int r = rr + p * 16;
      const float* src = nullptr;
      if (r < 65)                        src = Wq + (size_t)(head * 65 + r) * 1024;
      else if (r >= 80 && r < 144)       src = Wk + (size_t)(head * 64 + (r - 80)) * 1024;
      else if (r >= 144 && r < 208)      src = Wv + (size_t)(head * 64 + (r - 144)) * 1024;
      if (src) {
        store_bf16x4(&sm.l.Ws[r][cg * 4], *(const float4*)(src + k0 + cg * 4));
      } else {
        v2u z; z.x = 0; z.y = 0;
        *(v2u*)&sm.l.Ws[r][cg * 4] = z;
      }
    }
    if (k0 + 64 < 1024)  // hint next X chunk into cache (global_prefetch_b8)
      __builtin_prefetch(x + (size_t)(tokenBase + rr) * 1024 + k0 + 64, 0, 0);
    __syncthreads();

#pragma unroll
    for (int ks = 0; ks < 2; ++ks) {
      const v16bf a = load_frag(&sm.l.Xs[mtile * 16][ks * 32], lane);
      v16bf b[7];
#pragma unroll
      for (int t = 0; t < 7; ++t)
        b[t] = load_frag(&sm.l.Ws[(tBeg + t) * 16][ks * 32], lane);
#pragma unroll
      for (int t = 0; t < 7; ++t)
        acc[t] = __builtin_amdgcn_wmma_f32_16x16x32_bf16(
            false, a, false, b[t], (short)0, acc[t], false, false);
    }
    __syncthreads();
  }

  // ---- spill accumulators to Zs[token][col]  (C/D layout: lane=N, VGPR r=M%8)
  {
    const int hi = lane >> 4;
#pragma unroll
    for (int t = 0; t < 7; ++t) {
      const int col = (tBeg + t) * 16 + (lane & 15);
#pragma unroll
      for (int r = 0; r < 8; ++r)
        sm.Zs[mtile * 16 + r + 8 * hi][col] = acc[t][r];
    }
  }
  __syncthreads();

  // ---- per-token epilogue: q-last softmax slot, time-norm gate, k softmax, gate v
  if (tid < 64) {
    const int gt   = tokenBase + tid;
    const int lpos = gt & 4095;                       // l = 4096
    const float* z = &sm.Zs[tid][0];

    float m = z[0];
    for (int j = 1; j < 65; ++j) m = fmaxf(m, z[j]);
    float s = 0.f;
    for (int j = 0; j < 65; ++j) s += __expf(z[j] - m);
    const float qlast = __expf(z[64] - m) / s;

    // faithful sum(time^2) over 2*hd trig features (analytically == 2)
    const float delta = hdelt[head];
    float t2 = 0.f;
    for (int d = 0; d < 64; ++d) {
      float sn, cs;
      __sincosf(((float)lpos + delta) * ta[d], &sn, &cs);
      t2 += 2.f * (cs * cs + sn * sn);
    }
    t2 *= (1.f / 64.f);
    const float qkw = (1.f - qlast) * t2;

    float mk = z[80];
    for (int d = 1; d < 64; ++d) mk = fmaxf(mk, z[80 + d]);
    float sk = 0.f;
    for (int d = 0; d < 64; ++d) sk += __expf(z[80 + d] - mk);
    const float scale = amask[gt] * qkw / sk;

    unsigned short* g = G + (size_t)gt * 1024 + head * 64;
    for (int d = 0; d < 64; ++d)
      g[d] = f32_bf16(__expf(z[80 + d] - mk) * scale * z[144 + d]);
  }
}

// ---------------------------------------------------------------------------
// Kernel 2: out = g @ Wo^T + bo.  M=16384, N=1024, K=1024.
// 128x128 tile per WG, 8 waves (one 16-row M-stripe x 8 N-tiles each).
// G tile (pure bf16 copy) staged by the TDM when available; padding feature
// reproduces the 72-bf16 LDS row stride (pad 4 DWORDs after every 32 DWORDs).
// ---------------------------------------------------------------------------
#define HAVE_TDM (__has_builtin(__builtin_amdgcn_tensor_load_to_lds) && \
                  __has_builtin(__builtin_amdgcn_s_wait_tensorcnt))

__global__ __launch_bounds__(256) void out_gemm_kernel(
    const unsigned short* __restrict__ G, const float* __restrict__ Wo,
    const float* __restrict__ bo, float* __restrict__ out) {
  __shared__ __align__(16) unsigned short Gs[128][72];    // 18 KB
  __shared__ __align__(16) unsigned short Wos[128][72];   // 18 KB

  const int tid   = threadIdx.x;
  const int lane  = tid & 31;
  const int wv    = tid >> 5;
  const int mBase = (int)(blockIdx.x >> 3) * 128;
  const int nBase = (int)(blockIdx.x & 7) * 128;

  v8f acc[8] = {};

  for (int k0 = 0; k0 < 1024; k0 += 64) {
#if HAVE_TDM
    if (wv == 0) {
      // ---- D# descriptor (CDNA5 ISA sect.8): 2D bf16 tensor,
      // tile 64(K) x 128(rows), LDS pad 4 DWORDs per 32-DWORD row -> stride 72.
      const unsigned long long ga =
          (unsigned long long)(const void*)(G + (size_t)(mBase)*1024 + k0);
      const unsigned int lds_off = (unsigned int)(size_t)(void*)&Gs[0][0];
      v4u g0;
      g0.x = 1u;                                        // count=1 user D#
      g0.y = lds_off;                                   // lds_addr (bytes)
      g0.z = (unsigned int)ga;                          // global_addr[31:0]
      g0.w = (unsigned int)(ga >> 32) | (2u << 30);     // addr[56:32] | type=2
      v8i g1 = {0, 0, 0, 0, 0, 0, 0, 0};
      g1[0] = (1 << 16)        // data_size = 2 bytes
            | (1 << 20)        // pad_enable
            | (4 << 22)        // pad_interval: 32 DWORDs (one 64-bf16 row)
            | (3 << 25);       // pad_amount: 4 DWORDs (8 bf16)
      g1[1] = (int)((1024u & 0xFFFFu) << 16);               // tensor_dim0 lo16
      g1[2] = (int)((1024u >> 16) | ((16384u & 0xFFFFu) << 16)); // dim0 hi | dim1 lo
      g1[3] = (int)((16384u >> 16) | (64u << 16));          // dim1 hi | tile_dim0=64
      g1[4] = 128;                                          // tile_dim1=128
      g1[5] = 1024;                                         // tensor_dim0_stride lo32
      const v4i gz = {0, 0, 0, 0};                          // 2D: groups 2/3 unused
#if __clang_major__ >= 23
      const v8i gz8 = {0, 0, 0, 0, 0, 0, 0, 0};
      __builtin_amdgcn_tensor_load_to_lds(g0, g1, gz, gz, gz8, 0);
#else
      __builtin_amdgcn_tensor_load_to_lds(g0, g1, gz, gz, 0);
#endif
    }
#else
    {   // fallback: manual 16B vector copies, 8 chunks/row
      const int qc = tid & 7;
      const int rg = tid >> 3;            // 0..31
#pragma unroll
      for (int p = 0; p < 4; ++p) {
        const int row = rg + p * 32;
        const v4u v = *(const v4u*)(G + (size_t)(mBase + row) * 1024 + k0 + qc * 8);
        *(v4u*)&Gs[row][qc * 8] = v;
      }
    }
#endif
    {   // Wo tile (128x64), f32 -> bf16 (conversion: must stay on VALU path)
      const int cg = tid & 15;
      const int rw = tid >> 4;            // 0..15
#pragma unroll
      for (int p = 0; p < 8; ++p) {
        const int row = rw + p * 16;
        store_bf16x4(&Wos[row][cg * 4],
                     *(const float4*)(Wo + (size_t)(nBase + row) * 1024 + k0 + cg * 4));
      }
    }
#if HAVE_TDM
    if (wv == 0) __builtin_amdgcn_s_wait_tensorcnt(0);  // DMA done before publish
#endif
    __syncthreads();

#pragma unroll
    for (int ks = 0; ks < 2; ++ks) {
      const v16bf a = load_frag(&Gs[wv * 16][ks * 32], lane);
      v16bf b[8];
#pragma unroll
      for (int t = 0; t < 8; ++t)
        b[t] = load_frag(&Wos[t * 16][ks * 32], lane);
#pragma unroll
      for (int t = 0; t < 8; ++t)
        acc[t] = __builtin_amdgcn_wmma_f32_16x16x32_bf16(
            false, a, false, b[t], (short)0, acc[t], false, false);
    }
    __syncthreads();
  }

  const int hi = lane >> 4;
#pragma unroll
  for (int t = 0; t < 8; ++t) {
    const int col = nBase + t * 16 + (lane & 15);
    const float bias = bo[col];
#pragma unroll
    for (int r = 0; r < 8; ++r) {
      const int row = mBase + wv * 16 + r + 8 * hi;
      out[(size_t)row * 1024 + col] = acc[t][r] + bias;
    }
  }
}

// ---------------------------------------------------------------------------
extern "C" void kernel_launch(void* const* d_in, const int* in_sizes, int n_in,
                              void* d_out, int out_size, void* d_ws, size_t ws_size,
                              hipStream_t stream) {
  (void)in_sizes; (void)n_in; (void)out_size; (void)ws_size;
  const float* x   = (const float*)d_in[0];
  const float* am  = (const float*)d_in[1];
  const float* Wq  = (const float*)d_in[2];
  const float* Wk  = (const float*)d_in[3];
  const float* Wv  = (const float*)d_in[4];
  const float* Wo  = (const float*)d_in[5];
  const float* bo  = (const float*)d_in[6];
  const float* ta  = (const float*)d_in[7];
  const float* hd  = (const float*)d_in[8];

  unsigned short* G = (unsigned short*)d_ws;   // 16384 x 1024 bf16 = 32 MB

  proj_gate_kernel<<<4096, 256, 0, stream>>>(x, am, Wq, Wk, Wv, ta, hd, G);
  out_gemm_kernel<<<1024, 256, 0, stream>>>(G, Wo, bo, (float*)d_out);
}
